// DeepSeekV3Router_65816078844700
// MI455X (gfx1250) — matrix operand order repaired and checked by reading, not compiled, not run on hardware
//
#include <hip/hip_runtime.h>
#include <hip/hip_bf16.h>
#include <math.h>

// DeepSeek-V3 router for MI455X (gfx1250, wave32).
// Stage 0: convert+transpose router weights f32[D][E] -> bf16[E][D] (L2-resident).
// Stage 1: bf16 WMMA GEMM (f32 accum) + fused sigmoid. B-chunks staged to LDS by
//          the Tensor Data Mover (tensor_load_to_lds, padded rows); A-chunks
//          staged with inline f32->bf16 conversion (x read exactly once).
// Stage 2: per-token grouped top-k routing.

typedef __attribute__((ext_vector_type(16))) __bf16 v16bf;
typedef __attribute__((ext_vector_type(8)))  float  v8f;
typedef __attribute__((ext_vector_type(4)))  unsigned int v4u;
typedef __attribute__((ext_vector_type(8)))  int    v8i;
typedef __attribute__((ext_vector_type(4)))  int    v4i;

#define T_TOK 8192
#define D_DIM 7168
#define E_EXP 256
#define N_GRP 8
#define EPG   32
#define TOPG  4
#define TOPK  8
#define RSCALE 2.5f

#define KC     64                 // K-chunk per iteration
#define LDS_KP (KC + 8)           // bf16 row stride: 72 bf16 = 144B (16B aligned,
                                  // matches TDM pad: 32 dwords data + 4 dwords pad)

// ---------------------------------------------------------------------------
// Kernel 0: wgtT[e][k] = (bf16)wgt[k][e]   (32x32 tiles via LDS)
// ---------------------------------------------------------------------------
__global__ __launch_bounds__(256)
void wgt_transpose_bf16(const float* __restrict__ wgt, __bf16* __restrict__ wgtT) {
    __shared__ float tile[32][33];
    const int kb = blockIdx.x * 32;
    const int eb = blockIdx.y * 32;
    const int t  = threadIdx.x;
    #pragma unroll
    for (int i = 0; i < 4; ++i) {
        int id = t + i * 256;
        int r = id >> 5, c = id & 31;                 // r: k, c: e  (coalesced over e)
        tile[r][c] = wgt[(size_t)(kb + r) * E_EXP + (eb + c)];
    }
    __syncthreads();
    #pragma unroll
    for (int i = 0; i < 4; ++i) {
        int id = t + i * 256;
        int r = id >> 5, c = id & 31;                 // r: e, c: k
        wgtT[(size_t)(eb + r) * D_DIM + (kb + c)] = (__bf16)tile[c][r];
    }
}

// ---------------------------------------------------------------------------
// Kernel 1: scores = sigmoid(x @ W).  512 threads = 16 waves.
// Block tile: 32 tokens x 256 experts.  Wave w owns experts [w*16, w*16+16).
// ---------------------------------------------------------------------------
__global__ __launch_bounds__(512)
void router_gemm_sigmoid(const float* __restrict__ x,
                         const __bf16* __restrict__ wgtT,
                         float* __restrict__ scores) {
    // bf16 LDS tiles, 16B-aligned rows (stride 144B)
    __shared__ __bf16 Asb[32][LDS_KP];     //  32 tokens x 64 K     (~4.6 KB)
    __shared__ __bf16 Bsb[E_EXP][LDS_KP];  // 256 experts x 64 K    (~36 KB, TDM-filled)

    const int tid  = threadIdx.x;
    const int wave = tid >> 5;
    const int lane = tid & 31;
    const int half = lane >> 4;
    const int lr   = lane & 15;

    const int m0 = blockIdx.x * 32;
    const int n0 = wave * 16;

    v8f c0 = {};
    v8f c1 = {};

    // ---- TDM descriptor (B chunk: 256 rows x 64 bf16, row stride D, padded) ----
    const unsigned ldsB = (unsigned)(size_t)(&Bsb[0][0]);   // LDS byte offset
    // group1: data_size=1(2B) | pad_enable | pad_interval=4(32dw) | pad_amount=3(4dw)
    v8i g1;
    g1[0] = (int)((1u << 16) | (1u << 20) | (4u << 22) | (3u << 25));
    g1[1] = (int)(((unsigned)D_DIM & 0xFFFFu) << 16);       // tensor_dim0 lo16 @ [63:48]
    g1[2] = (int)(((unsigned)E_EXP & 0xFFFFu) << 16);       // dim0 hi16=0 | tensor_dim1 lo16
    g1[3] = (int)((unsigned)KC << 16);                      // dim1 hi16=0 | tile_dim0=64
    g1[4] = (int)(E_EXP);                                   // tile_dim1=256 | tile_dim2=0
    g1[5] = (int)(D_DIM);                                   // tensor_dim0_stride lo32
    g1[6] = 0;
    g1[7] = 0;
    const v4i gz4 = {0, 0, 0, 0};                           // 2D tensor: groups 2/3 unused
    const v8i gz8 = {0, 0, 0, 0, 0, 0, 0, 0};               // unused trailing group (6-arg form)

    for (int k0 = 0; k0 < D_DIM; k0 += KC) {
        __syncthreads();   // everyone done reading previous LDS tiles

        // ---- issue TDM for B chunk (wave 0 only; EXEC-independent DMA) ----
        if (wave == 0) {
            unsigned long long ga =
                (unsigned long long)(size_t)wgtT + (unsigned long long)k0 * 2ull;
            v4u g0;
            g0[0] = 1u;                                     // count=1, user mode
            g0[1] = ldsB;                                   // lds_addr
            g0[2] = (unsigned)(ga & 0xFFFFFFFFull);         // global_addr lo
            g0[3] = (unsigned)((ga >> 32) & 0x01FFFFFFull)  // global_addr hi (bits 56:32)
                  | (2u << 30);                             // type=2 (image)
            __builtin_amdgcn_tensor_load_to_lds(g0, g1, gz4, gz4, gz8, 0);
        }

        // ---- stage A chunk with f32->bf16 conversion (convert once, not per wave) ----
        // 32 rows x 32 pairs = 1024 pairs; 2 per thread; coalesced float2 reads.
        #pragma unroll
        for (int i = 0; i < 2; ++i) {
            int p  = tid + i * 512;
            int r  = p >> 5;            // token row
            int pp = p & 31;            // K pair index
            float2 f = *reinterpret_cast<const float2*>(
                x + (size_t)(m0 + r) * D_DIM + (k0 + 2 * pp));
            union { unsigned u; __bf16 h[2]; } pk;
            pk.h[0] = (__bf16)f.x;
            pk.h[1] = (__bf16)f.y;
            *reinterpret_cast<unsigned*>(&Asb[r][2 * pp]) = pk.u;
        }

        // speculative prefetch of next x chunk (gfx1250 global_prefetch_b8)
        if (k0 + KC < D_DIM) {
            __builtin_prefetch(x + (size_t)(m0 + (tid >> 4)) * D_DIM
                                 + (k0 + KC) + ((tid & 15) << 2), 0, 1);
        }

        if (wave == 0) __builtin_amdgcn_s_wait_tensorcnt(0);
        __syncthreads();   // B tile (TDM) + A tile (ds_store) visible to all waves

        // ---- two K-steps of 32, 2 WMMAs each ----
        #pragma unroll
        for (int kt = 0; kt < 2; ++kt) {
            // B fragment: 32 contiguous bf16 of expert row n0+lr
            union { uint4 q[2]; v16bf v; } ub;
            {
                const uint4* bp = reinterpret_cast<const uint4*>(
                    &Bsb[n0 + lr][kt * 32 + half * 16]);
                ub.q[0] = bp[0];
                ub.q[1] = bp[1];
            }
            // A fragments: 2 x 16B contiguous per M-tile (contiguous k = 16*half + s, same as B)
            union { uint4 q[2]; v16bf v; } ua0, ua1;
            ua0.q[0] = *reinterpret_cast<const uint4*>(&Asb[lr     ][kt * 32 + half * 16]);
            ua0.q[1] = *reinterpret_cast<const uint4*>(&Asb[lr     ][kt * 32 + half * 16 + 8]);
            ua1.q[0] = *reinterpret_cast<const uint4*>(&Asb[lr + 16][kt * 32 + half * 16]);
            ua1.q[1] = *reinterpret_cast<const uint4*>(&Asb[lr + 16][kt * 32 + half * 16 + 8]);

            c0 = __builtin_amdgcn_wmma_f32_16x16x32_bf16(
                    false, ua0.v, false, ub.v, (short)0, c0, false, false);
            c1 = __builtin_amdgcn_wmma_f32_16x16x32_bf16(
                    false, ua1.v, false, ub.v, (short)0, c1, false, false);
        }
    }

    // ---- fused sigmoid + store. C layout: VGPR r -> M = r + 8*half, N = lr.
    const int nIdx = n0 + lr;
    #pragma unroll
    for (int r = 0; r < 8; ++r) {
        int mA = m0 + r + 8 * half;
        int mB = mA + 16;
        scores[(size_t)mA * E_EXP + nIdx] = 1.0f / (1.0f + __expf(-c0[r]));
        scores[(size_t)mB * E_EXP + nIdx] = 1.0f / (1.0f + __expf(-c1[r]));
    }
}

// ---------------------------------------------------------------------------
// Kernel 2: grouped top-k routing. One thread per token (scores are L2 hot).
// ---------------------------------------------------------------------------
__global__ __launch_bounds__(256)
void router_topk(const float* __restrict__ scores,
                 const float* __restrict__ bias,
                 float* __restrict__ outW,
                 int* __restrict__ outI) {
    int t = blockIdx.x * blockDim.x + threadIdx.x;
    if (t >= T_TOK) return;
    const float* row = scores + (size_t)t * E_EXP;

    // group score = sum of top-2 of (score + bias) within each group of 32
    float gs[N_GRP];
    #pragma unroll
    for (int g = 0; g < N_GRP; ++g) {
        float m1 = -3.4e38f, m2 = -3.4e38f;
        for (int j = 0; j < EPG; ++j) {
            float v = row[g * EPG + j] + bias[g * EPG + j];
            if (v > m1) { m2 = m1; m1 = v; }
            else if (v > m2) { m2 = v; }
        }
        gs[g] = m1 + m2;
    }

    // top-4 groups -> bitmask
    unsigned gmask = 0;
    #pragma unroll
    for (int sel = 0; sel < TOPG; ++sel) {
        float best = -3.4e38f; int bg = 0;
        #pragma unroll
        for (int g = 0; g < N_GRP; ++g) {
            if (!((gmask >> g) & 1) && gs[g] > best) { best = gs[g]; bg = g; }
        }
        gmask |= 1u << bg;
    }

    // top-8 experts over masked s (non-selected groups forced to 0.0, as in ref)
    float wv[TOPK]; int wi[TOPK];
    #pragma unroll
    for (int k = 0; k < TOPK; ++k) { wv[k] = -3.4e38f; wi[k] = -1; }
    for (int e = 0; e < E_EXP; ++e) {
        float v = ((gmask >> (e >> 5)) & 1) ? (row[e] + bias[e]) : 0.0f;
        if (v > wv[TOPK - 1]) {             // strict >: earlier index wins ties
            int j = TOPK - 1;
            while (j > 0 && v > wv[j - 1]) {
                wv[j] = wv[j - 1]; wi[j] = wi[j - 1]; --j;
            }
            wv[j] = v; wi[j] = e;
        }
    }

    // gather original sigmoid scores, normalize, scale
    float w[TOPK]; float sum = 0.0f;
    #pragma unroll
    for (int k = 0; k < TOPK; ++k) { w[k] = row[wi[k]]; sum += w[k]; }
    float inv = RSCALE / (sum + 1e-20f);
    #pragma unroll
    for (int k = 0; k < TOPK; ++k) {
        outW[(size_t)t * TOPK + k] = w[k] * inv;
        outI[(size_t)t * TOPK + k] = wi[k];
    }
}

// ---------------------------------------------------------------------------
extern "C" void kernel_launch(void* const* d_in, const int* in_sizes, int n_in,
                              void* d_out, int out_size, void* d_ws, size_t ws_size,
                              hipStream_t stream) {
    const float* x    = (const float*)d_in[0];   // (T, D)
    const float* wgt  = (const float*)d_in[1];   // (D, E)
    const float* bias = (const float*)d_in[2];   // (E,)

    float* outW = (float*)d_out;                       // (T, 8) weights
    int*   outI = (int*)d_out + (size_t)T_TOK * TOPK;  // (T, 8) indices

    // workspace layout: scores (T*E f32 = 8 MB) | wgtT (E*D bf16 = 3.5 MB)
    float*  scores = (float*)d_ws;
    __bf16* wgtT   = (__bf16*)((char*)d_ws + (size_t)T_TOK * E_EXP * sizeof(float));

    wgt_transpose_bf16<<<dim3(D_DIM / 32, E_EXP / 32), dim3(256), 0, stream>>>(wgt, wgtT);
    router_gemm_sigmoid<<<dim3(T_TOK / 32), dim3(512), 0, stream>>>(x, wgtT, scores);
    router_topk<<<dim3(T_TOK / 256), dim3(256), 0, stream>>>(scores, bias, outW, outI);
}
